// CausalAttention_90692529422468
// MI455X (gfx1250) — compile-verified, matrix operand order checked by reading, not run on hardware
//
#include <hip/hip_runtime.h>
#include <hip/hip_bf16.h>

// ---------------------------------------------------------------------------
// CDNA5 (gfx1250) causal multi-head attention, B=4 N=2048 C=1024 H=16 D=64.
// All matmuls on v_wmma_f32_16x16x32_bf16 (wave32), fp32 accumulation.
// FlashAttention-style online softmax: the 2048x2048 score matrix never
// leaves the WGP.  bf16 tile staging into LDS is done by the Tensor Data
// Mover (tensor_load_to_lds + s_wait_tensorcnt), with the D# pad fields
// generating the bank-padding LDS row stride directly.
// ---------------------------------------------------------------------------

typedef __attribute__((ext_vector_type(16))) __bf16 v16bf;
typedef __attribute__((ext_vector_type(8)))  float  v8f;
typedef __attribute__((ext_vector_type(4)))  unsigned int v4u;
typedef __attribute__((ext_vector_type(8)))  int    v8i;
typedef __attribute__((ext_vector_type(4)))  int    v4i;

#define WMMA_BF16(A, B, C) \
  __builtin_amdgcn_wmma_f32_16x16x32_bf16(false, (A), false, (B), (short)0, (C), false, false)

__device__ __forceinline__ v8f vzero8() {
  v8f z = {0.f, 0.f, 0.f, 0.f, 0.f, 0.f, 0.f, 0.f};
  return z;
}

// ---------------------------------------------------------------------------
// Tensor Data Mover: 2-D bf16 tile load, global -> LDS, with LDS row padding.
// D# layout per CDNA5 ISA ch.8 (group0: flags/lds/global addr, group1: dims).
//   pad_interval_code: pad after 2^(code+1) dwords; pad_amount_code: code+1
//   dwords of pad.  data_size=1 (2-byte elements).
// This toolchain exposes the 6-arg builtin: (g0 u32x4, g1 i32x8, g2 i32x4,
// g3 i32x4, extra i32x8, cpol i32); 2-D tiles leave groups 2/3 zero.
// Issued once per wave (EXEC-ignored); caller gates to one wave and waits
// TENSORcnt==0 before the workgroup barrier releases consumers.
// ---------------------------------------------------------------------------
__device__ __forceinline__ void tdm_load_2d(
    unsigned lds_byte_addr, const void* gaddr,
    unsigned tensor_d0, unsigned tensor_d1, unsigned stride0,
    unsigned tile_d0, unsigned tile_d1,
    unsigned pad_interval_code, unsigned pad_amount_code)
{
  const unsigned long long ga = (unsigned long long)(uintptr_t)gaddr;
  v4u g0 = { 1u,                                   // count=1 valid descriptor
             lds_byte_addr,                        // LDS byte address
             (unsigned)(ga & 0xffffffffull),       // global_addr[31:0]
             (unsigned)((ga >> 32) & 0x01ffffffu)  // global_addr[56:32]
               | (2u << 30) };                     // type=2 ("image")
  v8i g1;
  g1[0] = (int)((1u << 16)                         // data_size: 2 bytes
              | (1u << 20)                         // pad_enable
              | (pad_interval_code << 22)
              | (pad_amount_code << 25));
  g1[1] = (int)((tensor_d0 & 0xffffu) << 16);      // tensor_dim0[15:0]
  g1[2] = (int)((tensor_d0 >> 16) | ((tensor_d1 & 0xffffu) << 16));
  g1[3] = (int)((tensor_d1 >> 16) | ((tile_d0 & 0xffffu) << 16));
  g1[4] = (int)(tile_d1 & 0xffffu);                // tile_dim1 (tile_dim2=0)
  g1[5] = (int)stride0;                            // tensor_dim0_stride[31:0]
  g1[6] = 0;
  g1[7] = 0;
  v4i z4 = {0, 0, 0, 0};
  v8i z8 = {0, 0, 0, 0, 0, 0, 0, 0};
  __builtin_amdgcn_tensor_load_to_lds(g0, g1, z4, z4, z8, 0);
}

// Load a 16-bit WMMA A/B fragment (16 rows/cols x 32 K) from an LDS tile row.
// CDNA5 16-bit A layout: lanes 0-15 hold K={0..7,16..23}, lanes 16-31 hold
// K={8..15,24..31} -> two contiguous 16B runs per lane = 2x ds_load_b128.
__device__ __forceinline__ v16bf ld_frag(const __bf16* rowptr, int lane) {
  const int off = (lane & 16) ? 8 : 0;
  v16bf f;
  uint4* fp = reinterpret_cast<uint4*>(&f);
  fp[0] = *reinterpret_cast<const uint4*>(rowptr + off);
  fp[1] = *reinterpret_cast<const uint4*>(rowptr + off + 16);
  return f;
}

// ---------------------------------------------------------------------------
// Kernel 1: QKV projection.  X[8192,1024] (f32) x Wqkv[3072,1024] (f32) ->
// bf16 Q,K in [B*H, N, D] and V transposed in [B*H, D, N].
// f32 sources need on-the-fly bf16 conversion, so these tiles stay on the
// VALU staging path; next-tile global_prefetch_b8 hides the latency.
// ---------------------------------------------------------------------------
__global__ __launch_bounds__(256) void ca_qkv_gemm(
    const float* __restrict__ X, const float* __restrict__ W,
    __bf16* __restrict__ qw, __bf16* __restrict__ kw, __bf16* __restrict__ vw)
{
  constexpr int LD = 40;                       // 80B row stride (16B multiple)
  __shared__ __align__(16) __bf16 sA[128 * LD];
  __shared__ __align__(16) __bf16 sB[128 * LD];

  const int tid  = threadIdx.x;
  const int lane = tid & 31;
  const int wave = tid >> 5;
  const int wr   = wave >> 1;                  // 0..3 -> 32-row group
  const int wc   = wave & 1;                   // 0..1 -> 64-col group
  const int m0   = blockIdx.y * 128;
  const int n0   = blockIdx.x * 128;

  v8f acc[2][4];
  for (int i = 0; i < 2; ++i)
    for (int j = 0; j < 4; ++j) acc[i][j] = vzero8();

  for (int k0 = 0; k0 < 1024; k0 += 32) {
    if (k0 + 32 < 1024) {                      // prefetch next K tile
      const int pr = tid >> 1, pc = (tid & 1) * 16;
      __builtin_prefetch(X + (size_t)(m0 + pr) * 1024 + k0 + 32 + pc, 0, 1);
      __builtin_prefetch(W + (size_t)(n0 + pr) * 1024 + k0 + 32 + pc, 0, 1);
    }
    __syncthreads();
    // Stage X tile (128x32 f32 -> bf16): 1024 float4, 4 per thread.
    for (int idx = tid; idx < 1024; idx += 256) {
      const int r = idx >> 3, c4 = idx & 7;
      const float4 f = *reinterpret_cast<const float4*>(
          X + (size_t)(m0 + r) * 1024 + k0 + c4 * 4);
      __bf16* d = &sA[r * LD + c4 * 4];
      d[0] = (__bf16)f.x; d[1] = (__bf16)f.y;
      d[2] = (__bf16)f.z; d[3] = (__bf16)f.w;
    }
    // Stage W tile (128 out-features x 32 k).
    for (int idx = tid; idx < 1024; idx += 256) {
      const int r = idx >> 3, c4 = idx & 7;
      const float4 f = *reinterpret_cast<const float4*>(
          W + (size_t)(n0 + r) * 1024 + k0 + c4 * 4);
      __bf16* d = &sB[r * LD + c4 * 4];
      d[0] = (__bf16)f.x; d[1] = (__bf16)f.y;
      d[2] = (__bf16)f.z; d[3] = (__bf16)f.w;
    }
    __syncthreads();

    v16bf af[2], bfr[4];
    for (int rt = 0; rt < 2; ++rt) {
      const int row = wr * 32 + rt * 16 + (lane & 15);
      af[rt] = ld_frag(&sA[row * LD], lane);
    }
    for (int ct = 0; ct < 4; ++ct) {
      const int col = wc * 64 + ct * 16 + (lane & 15);
      bfr[ct] = ld_frag(&sB[col * LD], lane);
    }
    for (int rt = 0; rt < 2; ++rt)
      for (int ct = 0; ct < 4; ++ct)
        acc[rt][ct] = WMMA_BF16(af[rt], bfr[ct], acc[rt][ct]);
  }

  // Branch-light scatter epilogue into attention-friendly bf16 layouts.
  const int hi = (lane & 16) ? 8 : 0;
  for (int rt = 0; rt < 2; ++rt) {
    for (int ct = 0; ct < 4; ++ct) {
      const int colg = n0 + wc * 64 + ct * 16 + (lane & 15);  // 0..3071
      const int part = colg >> 10;                            // 0=q 1=k 2=v
      const int rem  = colg & 1023;
      const int h    = rem >> 6;
      const int d    = rem & 63;
      __bf16* base = (part == 0) ? qw : (part == 1) ? kw : vw;
      for (int v = 0; v < 8; ++v) {
        const int rowg = m0 + wr * 32 + rt * 16 + v + hi;     // 0..8191
        const int b = rowg >> 11;
        const int n = rowg & 2047;
        const int bh = b * 16 + h;
        const size_t qk_idx = ((size_t)bh * 2048 + n) * 64 + d;
        const size_t vt_idx = ((size_t)bh * 64 + d) * 2048 + n;  // V^T
        base[(part == 2) ? vt_idx : qk_idx] = (__bf16)acc[rt][ct][v];
      }
    }
  }
}

// ---------------------------------------------------------------------------
// Kernel 2: causal flash attention.  grid=(N/64, B*H), block=128 (4 waves).
// Q/K/V^T tiles staged by the Tensor Data Mover (wave 0 issues, TENSORcnt
// waits, barrier releases).  pad codes (4,3): 32-dword rows + 4-dword pad
// -> 72-half LDS stride.
// ---------------------------------------------------------------------------
__global__ __launch_bounds__(128) void ca_flash_attn(
    const __bf16* __restrict__ qw, const __bf16* __restrict__ kw,
    const __bf16* __restrict__ vw, __bf16* __restrict__ ow)
{
  constexpr int LDQ = 72;                      // 144B row stride (16B multiple)
  __shared__ __align__(16) __bf16 sQ[64 * LDQ];
  __shared__ __align__(16) __bf16 sK[64 * LDQ];
  __shared__ __align__(16) __bf16 sVt[64 * LDQ];
  __shared__ __align__(16) __bf16 sP[64 * LDQ];

  const int tid   = threadIdx.x;
  const int lane  = tid & 31;
  const int wave  = tid >> 5;
  const int qb    = blockIdx.x;                // query block 0..31
  const int bh    = blockIdx.y;                // 0..63
  const int qBase = qb * 64;
  const int hi    = (lane & 16) ? 8 : 0;
  const float scale = 0.125f;                  // D^-0.5, D=64

  const size_t baseQK = (size_t)bh * 2048 * 64;

  // Stage the 64x64 Q tile once via TDM.
  if (wave == 0) {
    tdm_load_2d((unsigned)(uintptr_t)sQ, qw + baseQK + (size_t)qBase * 64,
                64, 64, 64, 64, 64, 4, 3);
    __builtin_amdgcn_s_wait_tensorcnt(0);
  }
  __syncthreads();

  v16bf qa[2];
  {
    const int row = wave * 16 + (lane & 15);
    qa[0] = ld_frag(&sQ[row * LDQ], lane);
    qa[1] = ld_frag(&sQ[row * LDQ + 32], lane);
  }

  float mrow[8], lrow[8];
  v8f acc[4];
  for (int v = 0; v < 8; ++v) { mrow[v] = -3.0e38f; lrow[v] = 0.f; }
  for (int t = 0; t < 4; ++t) acc[t] = vzero8();

  for (int kb = 0; kb <= qb; ++kb) {
    __syncthreads();  // previous iteration done reading sK/sVt/sP
    const int kBase = kb * 64;
    if (wave == 0) {
      // K tile: rows = key index, contiguous D.
      tdm_load_2d((unsigned)(uintptr_t)sK, kw + baseQK + (size_t)kBase * 64,
                  64, 64, 64, 64, 64, 4, 3);
      // V^T tile: rows = d, 64 contiguous n starting at kBase (row len 2048).
      tdm_load_2d((unsigned)(uintptr_t)sVt, vw + (size_t)bh * 64 * 2048 + kBase,
                  2048, 64, 2048, 64, 64, 4, 3);
      __builtin_amdgcn_s_wait_tensorcnt(0);
    }
    __syncthreads();

    // S = Q K^T  (wave: 16 q-rows x 64 k-cols, 8 wmma)
    v8f s[4];
    for (int t = 0; t < 4; ++t) {
      const int col = t * 16 + (lane & 15);           // key index (local)
      const v16bf b0 = ld_frag(&sK[col * LDQ], lane);
      const v16bf b1 = ld_frag(&sK[col * LDQ + 32], lane);
      s[t] = vzero8();
      s[t] = WMMA_BF16(qa[0], b0, s[t]);
      s[t] = WMMA_BF16(qa[1], b1, s[t]);
    }

    // Scale + causal mask + per-row max (C-layout: lane%16 = col, elem v = row).
    float rm[8];
    for (int v = 0; v < 8; ++v) rm[v] = -3.0e38f;
    for (int t = 0; t < 4; ++t) {
      const int col = kBase + t * 16 + (lane & 15);
      for (int v = 0; v < 8; ++v) {
        const int row = qBase + wave * 16 + v + hi;
        float x = s[t][v] * scale;
        x = (col > row) ? -3.0e38f : x;
        s[t][v] = x;
        rm[v] = fmaxf(rm[v], x);
      }
    }
    for (int v = 0; v < 8; ++v) {                     // reduce across 16 cols
      float m = rm[v];
      m = fmaxf(m, __shfl_xor(m, 1));
      m = fmaxf(m, __shfl_xor(m, 2));
      m = fmaxf(m, __shfl_xor(m, 4));
      m = fmaxf(m, __shfl_xor(m, 8));
      rm[v] = m;
    }
    // Online softmax rescale.
    for (int v = 0; v < 8; ++v) {
      const float mnew = fmaxf(mrow[v], rm[v]);
      const float corr = __expf(mrow[v] - mnew);
      mrow[v] = mnew;
      lrow[v] *= corr;
      for (int t = 0; t < 4; ++t) acc[t][v] *= corr;
    }
    float rs[8];
    for (int v = 0; v < 8; ++v) rs[v] = 0.f;
    for (int t = 0; t < 4; ++t)
      for (int v = 0; v < 8; ++v) {
        const float p = __expf(s[t][v] - mrow[v]);
        s[t][v] = p;
        rs[v] += p;
      }
    for (int v = 0; v < 8; ++v) {
      float r = rs[v];
      r += __shfl_xor(r, 1);
      r += __shfl_xor(r, 2);
      r += __shfl_xor(r, 4);
      r += __shfl_xor(r, 8);
      lrow[v] += r;
    }

    // Restage P (bf16) so it can be read back in A-fragment layout.
    for (int t = 0; t < 4; ++t)
      for (int v = 0; v < 8; ++v)
        sP[(wave * 16 + v + hi) * LDQ + t * 16 + (lane & 15)] = (__bf16)s[t][v];
    __syncthreads();

    v16bf pa[2];
    {
      const int prow = wave * 16 + (lane & 15);
      pa[0] = ld_frag(&sP[prow * LDQ], lane);
      pa[1] = ld_frag(&sP[prow * LDQ + 32], lane);
    }
    // O += P V  (B-fragments from transposed V: row = d, contiguous K = j)
    for (int t = 0; t < 4; ++t) {
      const int col = t * 16 + (lane & 15);           // d index
      const v16bf b0 = ld_frag(&sVt[col * LDQ], lane);
      const v16bf b1 = ld_frag(&sVt[col * LDQ + 32], lane);
      acc[t] = WMMA_BF16(pa[0], b0, acc[t]);
      acc[t] = WMMA_BF16(pa[1], b1, acc[t]);
    }
  }

  // Normalize and write attn output as bf16 [B, N, C] (C index = h*64 + d).
  const int b = bh >> 4, h = bh & 15;
  for (int t = 0; t < 4; ++t)
    for (int v = 0; v < 8; ++v) {
      const int row = qBase + wave * 16 + v + hi;     // n
      const float o = acc[t][v] / lrow[v];
      ow[((size_t)(b * 2048 + row)) * 1024 + h * 64 + t * 16 + (lane & 15)] = (__bf16)o;
    }
}

// ---------------------------------------------------------------------------
// Kernel 3: output projection. A[8192,1024] (bf16) x Wproj[1024,1024] (f32)
// + bias -> f32 out.  A tile (native bf16) staged by TDM; pad codes (3,3):
// 16-dword rows + 4-dword pad -> 40-half LDS stride.  W tile converted on
// the VALU path.
// ---------------------------------------------------------------------------
__global__ __launch_bounds__(256) void ca_proj_gemm(
    const __bf16* __restrict__ A, const float* __restrict__ W,
    const float* __restrict__ bias, float* __restrict__ out)
{
  constexpr int LD = 40;
  __shared__ __align__(16) __bf16 sA[128 * LD];
  __shared__ __align__(16) __bf16 sB[128 * LD];

  const int tid  = threadIdx.x;
  const int lane = tid & 31;
  const int wave = tid >> 5;
  const int wr   = wave >> 1;
  const int wc   = wave & 1;
  const int m0   = blockIdx.y * 128;
  const int n0   = blockIdx.x * 128;

  v8f acc[2][4];
  for (int i = 0; i < 2; ++i)
    for (int j = 0; j < 4; ++j) acc[i][j] = vzero8();

  for (int k0 = 0; k0 < 1024; k0 += 32) {
    __syncthreads();
    if (wave == 0) {                           // TDM: A tile 128x32 bf16
      tdm_load_2d((unsigned)(uintptr_t)sA, A + (size_t)m0 * 1024 + k0,
                  1024, 128, 1024, 32, 128, 3, 3);
    }
    // Stage W tile (f32 -> bf16) on the VALU path meanwhile.
    for (int idx = tid; idx < 1024; idx += 256) {
      const int r = idx >> 3, c4 = idx & 7;
      const float4 f = *reinterpret_cast<const float4*>(
          W + (size_t)(n0 + r) * 1024 + k0 + c4 * 4);
      __bf16* d = &sB[r * LD + c4 * 4];
      d[0] = (__bf16)f.x; d[1] = (__bf16)f.y;
      d[2] = (__bf16)f.z; d[3] = (__bf16)f.w;
    }
    if (wave == 0) __builtin_amdgcn_s_wait_tensorcnt(0);
    __syncthreads();

    v16bf af[2], bfr[4];
    for (int rt = 0; rt < 2; ++rt) {
      const int row = wr * 32 + rt * 16 + (lane & 15);
      af[rt] = ld_frag(&sA[row * LD], lane);
    }
    for (int ct = 0; ct < 4; ++ct) {
      const int col = wc * 64 + ct * 16 + (lane & 15);
      bfr[ct] = ld_frag(&sB[col * LD], lane);
    }
    for (int rt = 0; rt < 2; ++rt)
      for (int ct = 0; ct < 4; ++ct)
        acc[rt][ct] = WMMA_BF16(af[rt], bfr[ct], acc[rt][ct]);
  }

  const int hi = (lane & 16) ? 8 : 0;
  for (int rt = 0; rt < 2; ++rt)
    for (int ct = 0; ct < 4; ++ct) {
      const int colg = n0 + wc * 64 + ct * 16 + (lane & 15);
      const float bv = bias[colg];
      for (int v = 0; v < 8; ++v) {
        const int rowg = m0 + wr * 32 + rt * 16 + v + hi;
        out[(size_t)rowg * 1024 + colg] = acc[rt][ct][v] + bv;
      }
    }
}

// ---------------------------------------------------------------------------
extern "C" void kernel_launch(void* const* d_in, const int* in_sizes, int n_in,
                              void* d_out, int out_size, void* d_ws, size_t ws_size,
                              hipStream_t stream) {
  const float* x      = (const float*)d_in[0];   // [4,2048,1024]
  const float* w_qkv  = (const float*)d_in[1];   // [3072,1024]
  const float* w_proj = (const float*)d_in[2];   // [1024,1024]
  const float* b_proj = (const float*)d_in[3];   // [1024]
  float* out = (float*)d_out;                    // [4,2048,1024]

  // Workspace: bf16 Q,K [B*H,N,D], V^T [B*H,D,N], attn-out [B*N,C] = 67.1 MB.
  const size_t HEADSZ = (size_t)64 * 2048 * 64;  // elements per q/k/v buffer
  __bf16* qw = (__bf16*)d_ws;
  __bf16* kw = qw + HEADSZ;
  __bf16* vw = kw + HEADSZ;
  __bf16* aw = vw + HEADSZ;

  dim3 g1(3072 / 128, 8192 / 128);               // 24 x 64
  ca_qkv_gemm<<<g1, 256, 0, stream>>>(x, w_qkv, qw, kw, vw);

  dim3 g2(2048 / 64, 64);                        // 32 x (B*H)
  ca_flash_attn<<<g2, 128, 0, stream>>>(qw, kw, vw, aw);

  dim3 g3(1024 / 128, 8192 / 128);               // 8 x 64
  ca_proj_gemm<<<g3, 256, 0, stream>>>(aw, w_proj, b_proj, out);
}
